// AF_LSTM_78434692759654
// MI455X (gfx1250) — matrix operand order repaired and checked by reading, not compiled, not run on hardware
//
#include <hip/hip_runtime.h>

typedef __bf16 bf16;
typedef bf16  v16bf __attribute__((ext_vector_type(16)));
typedef float v8f   __attribute__((ext_vector_type(8)));

union AFrag { v16bf v; unsigned int u[8]; };
union BFrag { v16bf v; uint4 q[2]; };

// ---------------------------------------------------------------------------
// Fragment loaders (CDNA5 wave32 WMMA 16x16x32 bf16 layouts, ISA 7.12.2)
// A (16x32, MxK): lane row M = lane&15, group = lane>>4.
//   VGPR pair p holds K = group*16 + e, e = 2p.
// B (32x16, KxN): packed buffer stores per (ktile,ntile): 32 lanes x 16 elems
//   contiguous, K = (lane>>4)*16 + e, N = lane&15.
// C/D (16x16 f32): VGPR v -> M = v + 8*(lane>>4), N = lane&15.
// ---------------------------------------------------------------------------
__device__ __forceinline__ v16bf load_a_rowmajor(const bf16* A, int lda) {
  const int lane = threadIdx.x & 31;
  const int row  = lane & 15;
  const int grp  = lane >> 4;
  const bf16* base = A + (long long)row * lda + grp * 16;
  AFrag f;
#pragma unroll
  for (int p = 0; p < 8; ++p) {
    const int e = p * 2;
    const int k = e;   // 0,2,4,...,14
    f.u[p] = *(const unsigned int*)(base + k);
  }
  return f.v;
}

__device__ __forceinline__ v16bf load_b_packed(const bf16* tile) {
  const int lane = threadIdx.x & 31;
  BFrag f;
  const uint4* p = (const uint4*)(tile + lane * 16);
  f.q[0] = p[0];
  f.q[1] = p[1];
  return f.v;
}

// Async DMA of one 16-byte chunk: global -> LDS (CDNA5 ASYNCcnt path).
// lds_addr is the 32-bit LDS byte address (low 32 bits of the generic
// pointer to __shared__, per the flat-aperture truncation rule).
__device__ __forceinline__ void async_copy_b128(unsigned lds_addr,
                                                const bf16* gaddr) {
  asm volatile("global_load_async_to_lds_b128 %0, %1, off"
               :
               : "v"(lds_addr), "v"(gaddr)
               : "memory");
}

__device__ __forceinline__ float sigmoidf_(float x) {
  return 1.0f / (1.0f + __expf(-x));
}

// ---------------------------------------------------------------------------
// Pack a row-major f32 [K, Nsrc] weight into bf16 WMMA-B fragment layout of a
// destination with Ntot total columns, placed at column offset noff.
// ---------------------------------------------------------------------------
__global__ void pack_b_kernel(const float* __restrict__ W, int K, int Nsrc,
                              int Ntot, int noff, bf16* __restrict__ dst) {
  long long i = (long long)blockIdx.x * blockDim.x + threadIdx.x;
  const int Kt = K >> 5, NtS = Nsrc >> 4, NtT = Ntot >> 4;
  const long long total = (long long)Kt * NtS * 512;
  if (i >= total) return;
  const int e = (int)(i & 15);
  const int l = (int)((i >> 4) & 31);
  const long long t = i >> 9;
  const int ntl = (int)(t % NtS);
  const int kt  = (int)(t / NtS);
  const int k = kt * 32 + (l >> 4) * 16 + e;
  const int n = ntl * 16 + (l & 15);
  const float v = W[(long long)k * Nsrc + n];
  const int ntg = (noff >> 4) + ntl;
  dst[(((long long)kt * NtT + ntg) * 32 + l) * 16 + e] = (bf16)v;
}

__global__ void f32_to_bf16_kernel(const float* __restrict__ src,
                                   bf16* __restrict__ dst, long long n) {
  long long i = (long long)blockIdx.x * blockDim.x + threadIdx.x;
  if (i < n) dst[i] = (bf16)src[i];
}

__global__ void concat_bias_kernel(const float* b0, const float* b1,
                                   const float* b2, const float* b3,
                                   float* __restrict__ dst, int H) {
  int i = blockIdx.x * blockDim.x + threadIdx.x;
  if (i >= 4 * H) return;
  int g = i / H, h = i % H;
  const float* s = (g == 0) ? b0 : (g == 1) ? b1 : (g == 2) ? b2 : b3;
  dst[i] = s[h];
}

// exp(wbias[:T,:T]) -> bf16 row-major [T,T]
__global__ void exp_wbias_kernel(const float* __restrict__ wbias, int MAXT,
                                 bf16* __restrict__ ew, int T) {
  long long i = (long long)blockIdx.x * blockDim.x + threadIdx.x;
  if (i >= (long long)T * T) return;
  int r = (int)(i / T), c = (int)(i % T);
  ew[i] = (bf16)__expf(wbias[(long long)r * MAXT + c]);
}

// eK = exp(K), eKV = eK*V; scatter both into WMMA-B packed layout of a
// per-batch [T, 2H] matrix (cols [0,H)=eKV, [H,2H)=eK).
__global__ void aft_pack_kv_kernel(const float* __restrict__ Kf,
                                   const float* __restrict__ Vf,
                                   bf16* __restrict__ bpk, int T, int H,
                                   long long total) {
  long long i = (long long)blockIdx.x * blockDim.x + threadIdx.x;
  if (i >= total) return;
  const int h = (int)(i % H);
  const long long st = i / H;
  const int s = (int)(st % T);
  const int b = (int)(st / T);
  const float ek  = __expf(Kf[i]);
  const float ekv = ek * Vf[i];
  const int Ntiles = (2 * H) >> 4;
  const int Kt = T >> 5;
  const int kt = s >> 5;
  const int l_hi = ((s >> 4) & 1) * 16;
  const int e = s & 15;
  const long long base = (long long)b * Kt * Ntiles * 512;
  {
    const int col = h;
    const long long idx = base + (((long long)kt * Ntiles + (col >> 4)) * 32 +
                                  (l_hi + (col & 15))) * 16 + e;
    bpk[idx] = (bf16)ekv;
  }
  {
    const int col = H + h;
    const long long idx = base + (((long long)kt * Ntiles + (col >> 4)) * 32 +
                                  (l_hi + (col & 15))) * 16 + e;
    bpk[idx] = (bf16)ek;
  }
}

// Yt = sigmoid(Q) * num / den   (num/den from nd [b][T, 2H] row-major f32)
__global__ void yt_kernel(const float* __restrict__ Q,
                          const float* __restrict__ nd,
                          bf16* __restrict__ yt, int T, int H, long long total) {
  long long i = (long long)blockIdx.x * blockDim.x + threadIdx.x;
  if (i >= total) return;
  const int h = (int)(i % H);
  const long long bt = i / H;      // b*T + t
  const long long row = bt * (2 * H);
  const float num = nd[row + h];
  const float den = nd[row + H + h];
  yt[i] = (bf16)(sigmoidf_(Q[i]) * (num / den));
}

// ---------------------------------------------------------------------------
// Generic batched WMMA GEMM:  C[b] = A[b] @ Bpk[b] (+bias), bf16 in, f32 acc.
// Block = 256 threads = 8 waves. Block tile = 128(M) x 64(N).
// Wave w computes M-tile (blockIdx.y*8 + w) over the block's 4 N-tiles.
// B fragments for the block's N-tiles are staged cooperatively into LDS via
// async global->LDS DMA (double-buffered; 4 KB contiguous chunk per k-step),
// cutting B VMEM traffic 8x versus per-wave loads.
// ---------------------------------------------------------------------------
__global__ void __launch_bounds__(256)
gemm_bf16_wmma(const bf16* __restrict__ A, long long strideA,
               const bf16* __restrict__ Bpk, long long strideB,
               float* __restrict__ C, long long strideC,
               bf16* __restrict__ Cbf, long long strideCbf,
               const float* __restrict__ bias, int M, int N, int K) {
  __shared__ __align__(16) bf16 Bsh[2 * 2048];   // 2 x 4KB double buffer

  const int bz = blockIdx.z;
  A   += (long long)bz * strideA;
  Bpk += (long long)bz * strideB;
  if (C)   C   += (long long)bz * strideC;
  if (Cbf) Cbf += (long long)bz * strideCbf;

  const int tid  = threadIdx.x;
  const int wave = tid >> 5;
  const int lane = tid & 31;
  const int mt   = blockIdx.y * 8 + wave;
  const int nt0  = blockIdx.x * 4;
  const int Ktiles = K >> 5;
  const int Ntiles = N >> 4;

  const unsigned ldsBase = (unsigned)(uintptr_t)(&Bsh[0]) + (unsigned)tid * 16u;

  v8f acc[4] = {};
  const bf16* Arow = A + (long long)mt * 16 * K;

  // prologue: stage k-step 0 (each thread DMAs one b128 of the 4KB chunk)
  async_copy_b128(ldsBase, Bpk + ((long long)0 * Ntiles + nt0) * 512 + tid * 8);

  for (int kt = 0; kt < Ktiles; ++kt) {
    __syncthreads();   // all waves done reading the buffer we are about to fill
    if (kt + 1 < Ktiles) {
      const bf16* src = Bpk + ((long long)(kt + 1) * Ntiles + nt0) * 512 + tid * 8;
      async_copy_b128(ldsBase + (unsigned)(((kt + 1) & 1) * 4096), src);
      __builtin_prefetch(Arow + (kt + 1) * 32, 0, 3);
      asm volatile("s_wait_asynccnt 0x1" ::: "memory");  // k-step kt has landed
    } else {
      asm volatile("s_wait_asynccnt 0x0" ::: "memory");
    }
    __syncthreads();   // LDS writes from all threads visible

    const v16bf a = load_a_rowmajor(Arow + kt * 32, K);
    const bf16* bt = &Bsh[(kt & 1) * 2048];
#pragma unroll
    for (int j = 0; j < 4; ++j) {
      const v16bf bb = load_b_packed(bt + j * 512);
      acc[j] = __builtin_amdgcn_wmma_f32_16x16x32_bf16(
          false, a, false, bb, (short)0, acc[j], false, false);
    }
  }

  const int col0  = lane & 15;
  const int rbase = (lane >> 4) * 8;
#pragma unroll
  for (int j = 0; j < 4; ++j) {
    const int col = (nt0 + j) * 16 + col0;
    const float bv = bias ? bias[col] : 0.0f;
#pragma unroll
    for (int v = 0; v < 8; ++v) {
      const long long row = (long long)mt * 16 + rbase + v;
      const float val = acc[j][v] + bv;
      if (C)   C[row * N + col]   = val;
      if (Cbf) Cbf[row * N + col] = (bf16)val;
    }
  }
}

// ---------------------------------------------------------------------------
// Fused: filt=relu(att); afx=LN(att,g1,b1); prod=LN(filt*afx,g2,b2) -> bf16.
// One 256-thread block per (b,t) row, H == 256.
// ---------------------------------------------------------------------------
__device__ __forceinline__ float block_sum_256(float v, float* red) {
  const int h = threadIdx.x;
  red[h] = v;
  __syncthreads();
  for (int s = 128; s > 0; s >>= 1) {
    if (h < s) red[h] += red[h + s];
    __syncthreads();
  }
  const float r = red[0];
  __syncthreads();
  return r;
}

__global__ void __launch_bounds__(256)
ln_fused_kernel(const float* __restrict__ att,
                const float* __restrict__ g1, const float* __restrict__ b1,
                const float* __restrict__ g2, const float* __restrict__ b2,
                bf16* __restrict__ prod_bf, int H) {
  __shared__ float red[256];
  const long long row = blockIdx.x;
  const int h = threadIdx.x;
  const float a = att[row * H + h];

  const float mu = block_sum_256(a, red) / H;
  const float d  = a - mu;
  const float var = block_sum_256(d * d, red) / H;
  const float afx = d * rsqrtf(var + 1e-5f) * g1[h] + b1[h];
  const float y   = fmaxf(a, 0.0f) * afx;

  const float mu2 = block_sum_256(y, red) / H;
  const float d2  = y - mu2;
  const float var2 = block_sum_256(d2 * d2, red) / H;
  prod_bf[row * H + h] = (bf16)(d2 * rsqrtf(var2 + 1e-5f) * g2[h] + b2[h]);
}

// ---------------------------------------------------------------------------
// Persistent LSTM: single 1024-thread workgroup (32 waves on one WGP).
// Per step: gates = gates_x[:,t,:] + h @ Ucat  via WMMA (M padded 8->16),
// then pointwise cell update; h staged as bf16 in LDS for the next step.
// ---------------------------------------------------------------------------
__global__ void __launch_bounds__(1024)
lstm_kernel(const float* __restrict__ gates_x, const bf16* __restrict__ Upk,
            float* __restrict__ hidden, float* __restrict__ hT,
            float* __restrict__ cT, int T) {
  __shared__ bf16  hsh[16 * 256];   // padded h rows (8..15 stay zero), 8 KB
  __shared__ float gsh[8 * 1024];   // gates for valid batch rows, 32 KB

  const int tid = threadIdx.x;
  for (int i = tid; i < 16 * 256; i += 1024) hsh[i] = (bf16)0.0f;
  __syncthreads();

  const int wave = tid >> 5;
  const int lane = tid & 31;
  const int nt0  = wave * 2;            // 64 N-tiles / 32 waves
  const int col0 = lane & 15;
  const int rbase = (lane >> 4) * 8;

  // cell state: thread owns (b0,h0) and (b1,h1)
  const int b0 = tid >> 8,          h0 = tid & 255;
  const int b1 = (tid + 1024) >> 8, h1 = tid & 255;
  float c0 = 0.0f, c1 = 0.0f, hv0 = 0.0f, hv1 = 0.0f;

  for (int t = 0; t < T; ++t) {
    v8f acc[2] = {};
#pragma unroll
    for (int kt = 0; kt < 8; ++kt) {
      const v16bf a = load_a_rowmajor((const bf16*)hsh + kt * 32, 256);
      const bf16* bt = Upk + ((long long)kt * 64 + nt0) * 512;
#pragma unroll
      for (int j = 0; j < 2; ++j) {
        const v16bf bb = load_b_packed(bt + j * 512);
        acc[j] = __builtin_amdgcn_wmma_f32_16x16x32_bf16(
            false, a, false, bb, (short)0, acc[j], false, false);
      }
    }
    // gates (valid rows only) + precomputed input contribution
#pragma unroll
    for (int j = 0; j < 2; ++j) {
      const int col = (nt0 + j) * 16 + col0;
#pragma unroll
      for (int v = 0; v < 8; ++v) {
        const int row = rbase + v;
        if (row < 8)
          gsh[row * 1024 + col] =
              acc[j][v] + gates_x[((long long)row * T + t) * 1024 + col];
      }
    }
    __syncthreads();

    {
      const float ig = sigmoidf_(gsh[b0 * 1024 + h0]);
      const float fg = sigmoidf_(gsh[b0 * 1024 + 256 + h0]);
      const float gg = tanhf(gsh[b0 * 1024 + 512 + h0]);
      const float og = sigmoidf_(gsh[b0 * 1024 + 768 + h0]);
      c0 = fg * c0 + ig * gg;
      hv0 = og * tanhf(c0);
    }
    {
      const float ig = sigmoidf_(gsh[b1 * 1024 + h1]);
      const float fg = sigmoidf_(gsh[b1 * 1024 + 256 + h1]);
      const float gg = tanhf(gsh[b1 * 1024 + 512 + h1]);
      const float og = sigmoidf_(gsh[b1 * 1024 + 768 + h1]);
      c1 = fg * c1 + ig * gg;
      hv1 = og * tanhf(c1);
    }
    hidden[((long long)b0 * T + t) * 256 + h0] = hv0;
    hidden[((long long)b1 * T + t) * 256 + h1] = hv1;
    hsh[b0 * 256 + h0] = (bf16)hv0;
    hsh[b1 * 256 + h1] = (bf16)hv1;
    __syncthreads();
  }

  hT[b0 * 256 + h0] = hv0;
  hT[b1 * 256 + h1] = hv1;
  cT[b0 * 256 + h0] = c0;
  cT[b1 * 256 + h1] = c1;
}

// ---------------------------------------------------------------------------
extern "C" void kernel_launch(void* const* d_in, const int* in_sizes, int n_in,
                              void* d_out, int out_size, void* d_ws,
                              size_t ws_size, hipStream_t stream) {
  (void)in_sizes; (void)n_in; (void)out_size; (void)ws_size;
  const int B = 8, T = 2048, D = 256, H = 256, MAXT = 2048;
  const long long BT = (long long)B * T;

  const float* x    = (const float*)d_in[0];
  const float* fc_w = (const float*)d_in[1];
  const float* fc_b = (const float*)d_in[2];
  const float* q_w  = (const float*)d_in[3];
  const float* q_b  = (const float*)d_in[4];
  const float* k_w  = (const float*)d_in[5];
  const float* k_b  = (const float*)d_in[6];
  const float* v_w  = (const float*)d_in[7];
  const float* v_b  = (const float*)d_in[8];
  const float* p_w  = (const float*)d_in[9];
  const float* p_b  = (const float*)d_in[10];
  const float* wbias= (const float*)d_in[11];
  const float* n1_g = (const float*)d_in[12];
  const float* n1_b = (const float*)d_in[13];
  const float* n2_g = (const float*)d_in[14];
  const float* n2_b = (const float*)d_in[15];
  const float* w_i  = (const float*)d_in[16];
  const float* u_i  = (const float*)d_in[17];
  const float* b_i  = (const float*)d_in[18];
  const float* w_f  = (const float*)d_in[19];
  const float* u_f  = (const float*)d_in[20];
  const float* b_f  = (const float*)d_in[21];
  const float* w_c  = (const float*)d_in[22];
  const float* u_c  = (const float*)d_in[23];
  const float* b_c  = (const float*)d_in[24];
  const float* w_o  = (const float*)d_in[25];
  const float* u_o  = (const float*)d_in[26];
  const float* b_o  = (const float*)d_in[27];

  float* out    = (float*)d_out;
  float* hidden = out;
  float* hT     = out + BT * H;
  float* cT     = hT + (long long)B * H;

  // ---- workspace layout ----
  char* ws = (char*)d_ws;
  size_t off = 0;
  auto alloc = [&](size_t bytes) -> char* {
    char* p = ws + off;
    off = (off + bytes + 255) & ~(size_t)255;
    return p;
  };
  bf16* x_bf    = (bf16*)alloc(BT * D * 2);
  bf16* fcw_pk  = (bf16*)alloc((size_t)D * H * 2);
  bf16* qw_pk   = (bf16*)alloc((size_t)H * H * 2);
  bf16* kw_pk   = (bf16*)alloc((size_t)H * H * 2);
  bf16* vw_pk   = (bf16*)alloc((size_t)H * H * 2);
  bf16* pw_pk   = (bf16*)alloc((size_t)H * H * 2);
  bf16* wcat_pk = (bf16*)alloc((size_t)H * 4 * H * 2);
  bf16* ucat_pk = (bf16*)alloc((size_t)H * 4 * H * 2);
  float* bcat   = (float*)alloc(4 * H * 4);
  bf16* xf_bf   = (bf16*)alloc(BT * H * 2);
  float* Qf     = (float*)alloc(BT * H * 4);
  float* Kf     = (float*)alloc(BT * H * 4);
  float* Vf     = (float*)alloc(BT * H * 4);
  bf16* ew_bf   = (bf16*)alloc((size_t)T * T * 2);
  bf16* bpk     = (bf16*)alloc((size_t)B * T * 2 * H * 2);
  float* nd     = (float*)alloc((size_t)B * T * 2 * H * 4);
  bf16* yt_bf   = (bf16*)alloc(BT * H * 2);
  float* att    = (float*)alloc(BT * H * 4);
  bf16* prod_bf = (bf16*)alloc(BT * H * 2);
  // gates_x aliases the Qf..nd region (all dead by the time it is written)
  float* gates_x = Qf;

  const int TB = 256;
  auto blocks = [&](long long n) { return (unsigned)((n + TB - 1) / TB); };

  // ---- pack weights into WMMA-B fragment layout ----
  {
    const long long pk = (long long)(D / 32) * (H / 16) * 512;
    pack_b_kernel<<<blocks(pk), TB, 0, stream>>>(fc_w, D, H, H, 0, fcw_pk);
    pack_b_kernel<<<blocks(pk), TB, 0, stream>>>(q_w, H, H, H, 0, qw_pk);
    pack_b_kernel<<<blocks(pk), TB, 0, stream>>>(k_w, H, H, H, 0, kw_pk);
    pack_b_kernel<<<blocks(pk), TB, 0, stream>>>(v_w, H, H, H, 0, vw_pk);
    pack_b_kernel<<<blocks(pk), TB, 0, stream>>>(p_w, H, H, H, 0, pw_pk);
    pack_b_kernel<<<blocks(pk), TB, 0, stream>>>(w_i, H, H, 4 * H, 0,     wcat_pk);
    pack_b_kernel<<<blocks(pk), TB, 0, stream>>>(w_f, H, H, 4 * H, H,     wcat_pk);
    pack_b_kernel<<<blocks(pk), TB, 0, stream>>>(w_c, H, H, 4 * H, 2 * H, wcat_pk);
    pack_b_kernel<<<blocks(pk), TB, 0, stream>>>(w_o, H, H, 4 * H, 3 * H, wcat_pk);
    pack_b_kernel<<<blocks(pk), TB, 0, stream>>>(u_i, H, H, 4 * H, 0,     ucat_pk);
    pack_b_kernel<<<blocks(pk), TB, 0, stream>>>(u_f, H, H, 4 * H, H,     ucat_pk);
    pack_b_kernel<<<blocks(pk), TB, 0, stream>>>(u_c, H, H, 4 * H, 2 * H, ucat_pk);
    pack_b_kernel<<<blocks(pk), TB, 0, stream>>>(u_o, H, H, 4 * H, 3 * H, ucat_pk);
    concat_bias_kernel<<<blocks(4 * H), TB, 0, stream>>>(b_i, b_f, b_c, b_o,
                                                         bcat, H);
  }

  // ---- x -> bf16 ----
  f32_to_bf16_kernel<<<blocks(BT * D), TB, 0, stream>>>(x, x_bf, BT * D);

  // ---- xf = x @ fc_w + fc_b (bf16 out only) ----
  gemm_bf16_wmma<<<dim3(H / 64, (unsigned)(BT / 128), 1), 256, 0, stream>>>(
      x_bf, 0, fcw_pk, 0, nullptr, 0, xf_bf, 0, fc_b, (int)BT, H, D);

  // ---- Q, K, V projections (f32 out) ----
  gemm_bf16_wmma<<<dim3(H / 64, (unsigned)(BT / 128), 1), 256, 0, stream>>>(
      xf_bf, 0, qw_pk, 0, Qf, 0, nullptr, 0, q_b, (int)BT, H, H);
  gemm_bf16_wmma<<<dim3(H / 64, (unsigned)(BT / 128), 1), 256, 0, stream>>>(
      xf_bf, 0, kw_pk, 0, Kf, 0, nullptr, 0, k_b, (int)BT, H, H);
  gemm_bf16_wmma<<<dim3(H / 64, (unsigned)(BT / 128), 1), 256, 0, stream>>>(
      xf_bf, 0, vw_pk, 0, Vf, 0, nullptr, 0, v_b, (int)BT, H, H);

  // ---- AFT: ew = exp(wbias), Bpk = pack([eK*V | eK]) ----
  exp_wbias_kernel<<<blocks((long long)T * T), TB, 0, stream>>>(wbias, MAXT,
                                                                ew_bf, T);
  aft_pack_kv_kernel<<<blocks(BT * H), TB, 0, stream>>>(Kf, Vf, bpk, T, H,
                                                        BT * H);

  // ---- batched [T,T] @ [T,2H] -> nd ----
  {
    const long long sB = (long long)(T / 32) * (2 * H / 16) * 512;
    const long long sC = (long long)T * 2 * H;
    gemm_bf16_wmma<<<dim3(2 * H / 64, T / 128, B), 256, 0, stream>>>(
        ew_bf, 0, bpk, sB, nd, sC, nullptr, 0, nullptr, T, 2 * H, T);
  }

  // ---- Yt = sigmoid(Q) * num / den ----
  yt_kernel<<<blocks(BT * H), TB, 0, stream>>>(Qf, nd, yt_bf, T, H, BT * H);

  // ---- att = Yt @ p_w + p_b ----
  gemm_bf16_wmma<<<dim3(H / 64, (unsigned)(BT / 128), 1), 256, 0, stream>>>(
      yt_bf, 0, pw_pk, 0, att, 0, nullptr, 0, p_b, (int)BT, H, H);

  // ---- fused relu + double LayerNorm -> prod (bf16) ----
  ln_fused_kernel<<<(unsigned)BT, 256, 0, stream>>>(att, n1_g, n1_b, n2_g,
                                                    n2_b, prod_bf, H);

  // ---- gates_x = prod @ [Wi|Wf|Wc|Wo] + bcat ----
  gemm_bf16_wmma<<<dim3(4 * H / 64, (unsigned)(BT / 128), 1), 256, 0, stream>>>(
      prod_bf, 0, wcat_pk, 0, gates_x, 0, nullptr, 0, bcat, (int)BT, 4 * H, H);

  // ---- persistent recurrent LSTM ----
  lstm_kernel<<<1, 1024, 0, stream>>>(gates_x, ucat_pk, hidden, hT, cT, T);
}
